// SparseMoE_11029476016645
// MI455X (gfx1250) — compile-verified, matrix operand order checked by reading, not run on hardware
//
#include <hip/hip_runtime.h>

// ---------------------------------------------------------------------------
// Sparse MoE (top-2 of 16 experts), N=16384 D=1024 H=4096, for MI455X/gfx1250.
// bf16 WMMA (v_wmma_f32_16x16x32_bf16) with fp32 accumulation.
// x tiles are gathered into LDS by the Tensor Data Mover (gather-mode
// tensor_load_to_lds, TENSORcnt-tracked), with in-flight LDS padding chosen so
// ds_read_b128 of WMMA A-tiles is bank-conflict-free.
// ---------------------------------------------------------------------------

typedef __bf16 bf16;
typedef __bf16 v16bf __attribute__((ext_vector_type(16)));
typedef __bf16 v8bf  __attribute__((ext_vector_type(8)));
typedef __bf16 v4bf  __attribute__((ext_vector_type(4)));
typedef float  v8f   __attribute__((ext_vector_type(8)));
typedef float  v4f   __attribute__((ext_vector_type(4)));
typedef unsigned int u32x4 __attribute__((ext_vector_type(4)));
typedef int          i32x8 __attribute__((ext_vector_type(8)));
typedef int          i32x4 __attribute__((ext_vector_type(4)));

static constexpr int N_TOK = 16384;
static constexpr int D_DIM = 1024;
static constexpr int E_EXP = 16;
static constexpr int H_DIM = 4096;
static constexpr int TOPK  = 2;
static constexpr int TPB   = 64;                             // tokens per expert block
static constexpr int SLOT_MAX = N_TOK * TOPK + E_EXP * TPB;  // padded slot bound

// TDM gather writes 1024B half-rows + 16B pad -> 1040B chunks.
// 1040B = 260 dwords == 4 (mod 64): chunk j starts at bank 4j mod 64, and the
// hi-half offset of 16 chunks is bank-neutral, so the 16 rows of a WMMA A-tile
// read 16 disjoint 4-bank groups (conflict-free ds_read_b128), all 16B aligned.
static constexpr int CHUNK_B = 1040;
static constexpr int TILE_B  = 32 * CHUNK_B;   // 16 tokens x {lo,hi} half-rows

union ABPack { v16bf v; v8bf h[2]; };

__device__ __forceinline__ int rfl(int x) { return __builtin_amdgcn_readfirstlane(x); }

// ---- x: f32 -> bf16 -------------------------------------------------------
__global__ void cvt_x_kernel(const float* __restrict__ x, bf16* __restrict__ xb) {
  size_t i = ((size_t)blockIdx.x * blockDim.x + threadIdx.x) * 4;
  v4f v = *(const v4f*)(x + i);
  v4bf o;
  o[0] = (bf16)v[0]; o[1] = (bf16)v[1]; o[2] = (bf16)v[2]; o[3] = (bf16)v[3];
  *(v4bf*)(xb + i) = o;
}

// ---- W1 [E][D][H] f32 -> W1T [E][H][D] bf16 (LDS-tiled transpose) ---------
__global__ void cvt_w1_kernel(const float* __restrict__ W1, bf16* __restrict__ w1t) {
  __shared__ float tile[32][33];
  int e = blockIdx.z;
  int h0 = blockIdx.x * 32, d0 = blockIdx.y * 32;
  int tx = threadIdx.x, ty = threadIdx.y;
#pragma unroll
  for (int i = 0; i < 4; ++i) {
    int d = d0 + ty + i * 8;
    tile[ty + i * 8][tx] = W1[((size_t)e * D_DIM + d) * H_DIM + h0 + tx];
  }
  __syncthreads();
#pragma unroll
  for (int i = 0; i < 4; ++i) {
    int h = h0 + ty + i * 8;
    w1t[((size_t)e * H_DIM + h) * D_DIM + d0 + tx] = (bf16)tile[tx][ty + i * 8];
  }
}

// ---- Wn [D][E] f32 -> WnT [E][D] bf16 -------------------------------------
__global__ void cvt_wn_kernel(const float* __restrict__ Wn, bf16* __restrict__ wnt) {
  int d = blockIdx.x * blockDim.x + threadIdx.x;
  if (d >= D_DIM) return;
#pragma unroll
  for (int e = 0; e < E_EXP; ++e)
    wnt[e * D_DIM + d] = (bf16)Wn[(size_t)d * E_EXP + e];
}

// ---- zero-init output / counters / slot arrays ----------------------------
__global__ void init_kernel(float* __restrict__ out, int* __restrict__ counts,
                            int* __restrict__ cursors, int* __restrict__ slot_tok,
                            float* __restrict__ slot_gate) {
  int i = blockIdx.x * blockDim.x + threadIdx.x;
  if (i < SLOT_MAX) { slot_tok[i] = 0; slot_gate[i] = 0.f; }
  if (i < N_TOK)    out[i] = 0.f;
  if (i < E_EXP)    { counts[i] = 0; cursors[i] = 0; }
}

// ---- router: logits = x @ Wn + bn, one wave per 16-token tile -------------
__global__ void router_kernel(const bf16* __restrict__ xb, const bf16* __restrict__ wnt,
                              const float* __restrict__ bn, float* __restrict__ logits) {
  int wid  = (int)((blockIdx.x * blockDim.x + threadIdx.x) >> 5);
  int lane = threadIdx.x & 31;
  int tok0 = wid * 16;
  int n16  = lane & 15;
  int sel8 = (lane & 16) ? 8 : 0;
  int mhi  = (lane & 16) ? 8 : 0;

  const bf16* arow = xb  + (size_t)(tok0 + n16) * D_DIM + sel8;
  const bf16* brow = wnt + (size_t)n16 * D_DIM + sel8;

  v8f acc = {0.f, 0.f, 0.f, 0.f, 0.f, 0.f, 0.f, 0.f};
  for (int d0 = 0; d0 < D_DIM; d0 += 32) {
    ABPack a, b;
    a.h[0] = *(const v8bf*)(arow + d0);
    a.h[1] = *(const v8bf*)(arow + d0 + 16);
    b.h[0] = *(const v8bf*)(brow + d0);
    b.h[1] = *(const v8bf*)(brow + d0 + 16);
    acc = __builtin_amdgcn_wmma_f32_16x16x32_bf16(false, a.v, false, b.v,
                                                  (short)0, acc, false, false);
  }
  float bnv = bn[n16];
#pragma unroll
  for (int v = 0; v < 8; ++v)
    logits[(size_t)(tok0 + v + mhi) * E_EXP + n16] = acc[v] + bnv;
}

// ---- top-2 + softmax over the two kept logits; count per expert -----------
__global__ void topk_kernel(const float* __restrict__ logits, int* __restrict__ tok_idx,
                            float* __restrict__ tok_gate, int* __restrict__ counts) {
  int n = blockIdx.x * blockDim.x + threadIdx.x;
  const float* row = logits + (size_t)n * E_EXP;
  float v0 = -3.4e38f; int i0 = 0;
#pragma unroll
  for (int e = 0; e < E_EXP; ++e) { float v = row[e]; if (v > v0) { v0 = v; i0 = e; } }
  float v1 = -3.4e38f; int i1 = 0;
#pragma unroll
  for (int e = 0; e < E_EXP; ++e) { float v = row[e]; if (e != i0 && v > v1) { v1 = v; i1 = e; } }
  float p1  = __expf(v1 - v0);           // softmax over {v0, v1}, max-shifted
  float inv = 1.f / (1.f + p1);
  tok_idx[2 * n]     = i0; tok_idx[2 * n + 1]  = i1;
  tok_gate[2 * n]    = inv; tok_gate[2 * n + 1] = p1 * inv;
  atomicAdd(&counts[i0], 1);
  atomicAdd(&counts[i1], 1);
}

// ---- per-expert segment bases, padded to TPB ------------------------------
__global__ void segbase_kernel(const int* __restrict__ counts, int* __restrict__ seg_base) {
  if (threadIdx.x == 0) {
    int acc = 0;
    for (int e = 0; e < E_EXP; ++e) {
      seg_base[e] = acc;
      acc += ((counts[e] + TPB - 1) / TPB) * TPB;
    }
    seg_base[E_EXP] = acc;
  }
}

// ---- scatter (token, gate) into expert-sorted slots -----------------------
__global__ void scatter_kernel(const int* __restrict__ tok_idx, const float* __restrict__ tok_gate,
                               const int* __restrict__ seg_base, int* __restrict__ cursors,
                               int* __restrict__ slot_tok, float* __restrict__ slot_gate) {
  int n = blockIdx.x * blockDim.x + threadIdx.x;
#pragma unroll
  for (int kk = 0; kk < TOPK; ++kk) {
    int e = tok_idx[2 * n + kk];
    int p = seg_base[e] + atomicAdd(&cursors[e], 1);
    slot_tok[p]  = n;
    slot_gate[p] = tok_gate[2 * n + kk];
  }
}

// ---- fused expert GEMM: out[tok] += g * (relu(x@W1[e]+b1[e]) @ W2[e] + b2[e])
// Block = 64 tokens (4 WMMA M-tiles) of ONE expert; 8 waves partition H.
// x rows are gathered into LDS by TDM gather-mode descriptors (16 half-rows
// each, 16B pad per 1024B row), then each wave streams W1T B-tiles from L2
// and reuses them across all 4 token tiles.
__global__ void __launch_bounds__(256)
expert_kernel(const bf16* __restrict__ xb, const bf16* __restrict__ w1t,
              const float* __restrict__ b1, const float* __restrict__ W2,
              const float* __restrict__ b2, const int* __restrict__ seg_base,
              const int* __restrict__ slot_tok, const float* __restrict__ slot_gate,
              float* __restrict__ out) {
  __shared__ __align__(16) char lds_x[4 * TILE_B];   // ~130 KB of 320 KB WGP LDS
  __shared__ int   s_tok[TPB];
  __shared__ float s_gate[TPB];
  __shared__ float s_out[TPB];

  int slot0 = blockIdx.x * TPB;
  if (slot0 >= seg_base[E_EXP]) return;    // uniform early-out past padded total
  int e = 0;
#pragma unroll
  for (int i = 1; i < E_EXP; ++i) if (slot0 >= seg_base[i]) e = i;

  int t = threadIdx.x;
  if (t < TPB) {
    s_tok[t]  = slot_tok[slot0 + t];
    s_gate[t] = slot_gate[slot0 + t];
    s_out[t]  = 0.f;
  }
  __syncthreads();

  if (t < 32) {
    // Wave 0 issues 8 TDM gather descriptors: per 16-token tile, one for the
    // lo 1024B half-rows (LDS chunks 0..15) and one for the hi (chunks 16..31).
    // xb is viewed as 32768 rows of 1024B; token tk -> rows 2tk, 2tk+1.
    unsigned ldsbase = (unsigned)(size_t)(void*)lds_x;         // LDS byte offset
    unsigned long long ga = (unsigned long long)(size_t)xb;
    int galo = rfl((int)(unsigned)ga);
    int gahi = rfl((int)(unsigned)(ga >> 32));

    i32x8 g1;
    g1[0] = 0x07D30000;          // data_size=8B | pad_en | interval=1024B | pad=16B
    g1[1] = (int)(128u << 16);   // tensor_dim0 = 128 (8B units)
    g1[2] = (int)0x80000000u;    // tensor_dim1 = 32768 (lo16 in [31:16])
    g1[3] = (int)(128u << 16);   // tensor_dim1 hi16 = 0 | tile_dim0 = 128
    g1[4] = 16;                  // tile_dim1 = 16 gather indices
    g1[5] = 128;                 // tensor_dim0_stride = 128 (8B units = 1024B)
    g1[6] = 0;
    g1[7] = 0;

    i32x8 gext = {0, 0, 0, 0, 0, 0, 0, 0};   // clang-23 6-arg form: extra group,
                                             // zero-filled (probe-verified call)

    for (int tt = 0; tt < 4; ++tt) {
      int idx[16];
#pragma unroll
      for (int j = 0; j < 16; ++j) idx[j] = s_tok[tt * 16 + j] * 2;
      for (int hlf = 0; hlf < 2; ++hlf) {
        u32x4 g0;
        g0[0] = 0x80000001u;     // count=1 | gather_mode=1 | 16-bit indices
        g0[1] = (unsigned)rfl((int)(ldsbase + tt * TILE_B + hlf * 16 * CHUNK_B));
        g0[2] = (unsigned)galo;
        g0[3] = ((unsigned)gahi & 0x01ffffffu) | 0x80000000u;  // type=2 (image)
        i32x4 g2, g3;
#pragma unroll
        for (int j = 0; j < 4; ++j) {
          g2[j] = rfl((idx[2 * j] + hlf)     | ((idx[2 * j + 1] + hlf) << 16));
          g3[j] = rfl((idx[8 + 2 * j] + hlf) | ((idx[9 + 2 * j] + hlf) << 16));
        }
        __builtin_amdgcn_tensor_load_to_lds(g0, g1, g2, g3, gext, 0);
      }
    }
    __builtin_amdgcn_s_wait_tensorcnt(0);
  }
  __syncthreads();

  int w = t >> 5, lane = t & 31;
  int n16  = lane & 15;
  int sel8 = (lane & 16) ? 8 : 0;
  int mhi  = (lane & 16) ? 8 : 0;

  float pacc[4][8];
#pragma unroll
  for (int tt = 0; tt < 4; ++tt)
#pragma unroll
    for (int v = 0; v < 8; ++v) pacc[tt][v] = 0.f;

  for (int hh = 0; hh < 32; ++hh) {
    int h = (w * 32 + hh) * 16 + n16;                 // this lane's H column
    const bf16* brow = w1t + ((size_t)e * H_DIM + h) * D_DIM + sel8;

    v8f acc[4] = {};
    for (int d0 = 0; d0 < D_DIM; d0 += 32) {
      ABPack b;
      b.h[0] = *(const v8bf*)(brow + d0);
      b.h[1] = *(const v8bf*)(brow + d0 + 16);
      __builtin_prefetch(brow + d0 + 512, 0, 0);      // global_prefetch_b8
      int dA = d0 + sel8;                             // K-chunk starts (halves)
      int dB = dA + 16;
#pragma unroll
      for (int tt = 0; tt < 4; ++tt) {
        const char* abase = lds_x + tt * TILE_B;
        ABPack a;                                     // chunk = row + 16*(d>=512)
        a.h[0] = *(const v8bf*)(abase + (n16 + ((dA >> 9) << 4)) * CHUNK_B + (dA & 511) * 2);
        a.h[1] = *(const v8bf*)(abase + (n16 + ((dB >> 9) << 4)) * CHUNK_B + (dB & 511) * 2);
        acc[tt] = __builtin_amdgcn_wmma_f32_16x16x32_bf16(
            false, a.v, false, b.v, (short)0, acc[tt], false, false);
      }
    }
    // fused epilogue: relu(+b1) then fold this 16-col tile into per-token scalar
    float b1v = b1[(size_t)e * H_DIM + h];
    float w2v = W2[(size_t)e * H_DIM + h];
#pragma unroll
    for (int tt = 0; tt < 4; ++tt)
#pragma unroll
      for (int v = 0; v < 8; ++v) {
        float hv = acc[tt][v] + b1v;
        hv = hv > 0.f ? hv : 0.f;
        pacc[tt][v] += hv * w2v;
      }
  }

  // cross-lane/cross-wave reduction: lane l, VGPR v maps to token tt*16+v+mhi
#pragma unroll
  for (int tt = 0; tt < 4; ++tt)
#pragma unroll
    for (int v = 0; v < 8; ++v)
      atomicAdd(&s_out[tt * 16 + v + mhi], pacc[tt][v]);
  __syncthreads();

  if (t < TPB) {
    float val = s_out[t] + b2[e];
    atomicAdd(&out[s_tok[t]], s_gate[t] * val);       // padding slots: gate==0
  }
}

// ---------------------------------------------------------------------------
extern "C" void kernel_launch(void* const* d_in, const int* in_sizes, int n_in,
                              void* d_out, int out_size, void* d_ws, size_t ws_size,
                              hipStream_t stream) {
  (void)in_sizes; (void)n_in; (void)out_size; (void)ws_size;
  const float* x  = (const float*)d_in[0];
  // d_in[1]=Wr, d_in[2]=br are dead in the reference (noise-only router)
  const float* Wn = (const float*)d_in[3];
  const float* bn = (const float*)d_in[4];
  const float* W1 = (const float*)d_in[5];
  const float* b1 = (const float*)d_in[6];
  const float* W2 = (const float*)d_in[7];
  const float* b2 = (const float*)d_in[8];
  float* out = (float*)d_out;

  char* ws = (char*)d_ws;
  size_t off = 0;
  auto alloc = [&](size_t bytes) -> char* {
    char* p = ws + off;
    off = (off + bytes + 255) & ~(size_t)255;
    return p;
  };
  bf16*  xb        = (bf16*) alloc((size_t)N_TOK * D_DIM * sizeof(bf16));
  bf16*  w1t       = (bf16*) alloc((size_t)E_EXP * H_DIM * D_DIM * sizeof(bf16));
  bf16*  wnt       = (bf16*) alloc((size_t)E_EXP * D_DIM * sizeof(bf16));
  float* logits    = (float*)alloc((size_t)N_TOK * E_EXP * sizeof(float));
  int*   tok_idx   = (int*)  alloc((size_t)N_TOK * TOPK * sizeof(int));
  float* tok_gate  = (float*)alloc((size_t)N_TOK * TOPK * sizeof(float));
  int*   counts    = (int*)  alloc(E_EXP * sizeof(int));
  int*   cursors   = (int*)  alloc(E_EXP * sizeof(int));
  int*   seg_base  = (int*)  alloc((E_EXP + 1) * sizeof(int));
  int*   slot_tok  = (int*)  alloc((size_t)SLOT_MAX * sizeof(int));
  float* slot_gate = (float*)alloc((size_t)SLOT_MAX * sizeof(float));

  // 1) precision conversion / weight transposes
  cvt_x_kernel <<<(N_TOK * D_DIM / 4) / 256, 256, 0, stream>>>(x, xb);
  cvt_w1_kernel<<<dim3(H_DIM / 32, D_DIM / 32, E_EXP), dim3(32, 8), 0, stream>>>(W1, w1t);
  cvt_wn_kernel<<<(D_DIM + 255) / 256, 256, 0, stream>>>(Wn, wnt);
  init_kernel  <<<(SLOT_MAX + 255) / 256, 256, 0, stream>>>(out, counts, cursors,
                                                            slot_tok, slot_gate);
  // 2) router (WMMA), 4 waves/block, one 16-token tile per wave
  router_kernel<<<N_TOK / 64, 128, 0, stream>>>(xb, wnt, bn, logits);
  // 3) routing decisions
  topk_kernel   <<<N_TOK / 256, 256, 0, stream>>>(logits, tok_idx, tok_gate, counts);
  segbase_kernel<<<1, 32, 0, stream>>>(counts, seg_base);
  scatter_kernel<<<N_TOK / 256, 256, 0, stream>>>(tok_idx, tok_gate, seg_base,
                                                  cursors, slot_tok, slot_gate);
  // 4) fused expert compute over padded slots (worst case bound, uniform exit)
  expert_kernel<<<SLOT_MAX / TPB, 256, 0, stream>>>(xb, w1t, b1, W2, b2, seg_base,
                                                    slot_tok, slot_gate, out);
}
